// SparseConv3d_39393440038985
// MI455X (gfx1250) — matrix operand order repaired and checked by reading, not compiled, not run on hardware
//
#include <hip/hip_runtime.h>
#include <hip/hip_bf16.h>

// ---------------------------------------------------------------------------
// Submanifold sparse 3x3x3 conv, 128->128 ch, 65536 voxels, 27 offsets.
// v2.1: block-cooperative weight staging through LDS (double-buffered, async
// global->LDS loads), 2 M-tiles per wave so every LDS B-fragment feeds two
// v_wmma_f32_16x16x32_bf16 ops. Conflict-free output via an inverted
// per-output gather table (no atomics).
// ---------------------------------------------------------------------------

#define NV      65536   // active voxels
#define KOFF    27      // kernel offsets
#define CCH     128     // C_IN == C_OUT
#define TSTRIDE 32      // gather-table row stride (27 padded to 32)
#define FRAG_K_BYTES 32768   // per-offset B fragments: 4 chunks * 8 ntiles * 1KB

typedef __bf16 bf16;
typedef __attribute__((ext_vector_type(16))) __bf16 v16bf;
typedef __attribute__((ext_vector_type(8)))  __bf16 v8bf;
typedef __attribute__((ext_vector_type(8)))  float  v8f;
typedef __attribute__((ext_vector_type(4)))  int    v4i;

// address-space-qualified views for the async global->LDS builtin
typedef __attribute__((address_space(1))) v4i gv4i;   // global
typedef __attribute__((address_space(3))) v4i sv4i;   // LDS

union ABu { v16bf v; v8bf h[2]; };

#if defined(__has_builtin)
#if __has_builtin(__builtin_amdgcn_global_load_async_to_lds_b128) && \
    __has_builtin(__builtin_amdgcn_s_wait_asynccnt)
#define USE_ASYNC_LDS 1
#endif
#endif

// ---------------------------------------------------------------------------
// Prep 1: features fp32 -> bf16
// ---------------------------------------------------------------------------
__global__ void feat_to_bf16_k(const float* __restrict__ f, bf16* __restrict__ fb) {
    int i = blockIdx.x * blockDim.x + threadIdx.x;   // grid sized exactly
    fb[i] = (bf16)f[i];
}

// ---------------------------------------------------------------------------
// Prep 2a/2b: dense per-output gather table (inverted rulebook)
// ---------------------------------------------------------------------------
__global__ void table_init_k(int* __restrict__ table) {
    int i = blockIdx.x * blockDim.x + threadIdx.x;
    table[i] = -1;
}
__global__ void table_fill_k(const int* __restrict__ in_idx,
                             const int* __restrict__ out_idx,
                             int* __restrict__ table) {
    int i = blockIdx.x * blockDim.x + threadIdx.x;   // i = k*NV + p
    int o = out_idx[i];
    if (o < NV) {
        int k = i >> 16;                             // NV = 2^16
        table[o * TSTRIDE + k] = in_idx[i];
    }
}

// ---------------------------------------------------------------------------
// Prep 3: repack weights fp32 -> bf16 B-fragments, LDS-bank-friendly order.
// Fragment (1KB) = [half(2)][lane(32)][e8(8)] bf16:
//   first 512B  : per-lane 16B = B elems e=0..7
//   second 512B : per-lane 16B = B elems e=8..15
// B 32x16 as packed here (contiguous K order, same as the A loads): lane l -> col n = l&15, K = (l>>4)*16 + e.
// Fragment order: [k][chunk(4)][ntile(8)].
// ---------------------------------------------------------------------------
__global__ void weight_prep_k(const float* __restrict__ w, bf16* __restrict__ frag) {
    int i = blockIdx.x * blockDim.x + threadIdx.x;   // total = 27*2^14 = 442368
    if (i >= KOFF * 16384) return;
    int e8    =  i        & 7;
    int lane  = (i >> 3)  & 31;
    int half  = (i >> 8)  & 1;
    int t     = (i >> 9)  & 7;
    int chunk = (i >> 12) & 3;
    int k     =  i >> 14;
    int e  = half * 8 + e8;
    int n  = t * 16 + (lane & 15);
    int kk = chunk * 32 + (lane >> 4) * 16 + e;      // C_IN index
    frag[i] = (bf16)w[((size_t)k * CCH + kk) * CCH + n];
}

// ---------------------------------------------------------------------------
// Stage one offset's 32KB of B fragments into LDS (256 threads, 16B units).
// Async global->LDS path (ASYNCcnt-tracked) when available.
// ---------------------------------------------------------------------------
__device__ __forceinline__ void stage_frag(const v4i* __restrict__ src, v4i* dst, int tid) {
#ifdef USE_ASYNC_LDS
#pragma unroll
    for (int j = 0; j < 8; ++j) {
        const int u = j * 256 + tid;
        __builtin_amdgcn_global_load_async_to_lds_b128(
            (gv4i*)(src + u),        // global source (AS1, const cast away)
            (sv4i*)(dst + u),        // LDS destination (AS3)
            0, 0);
    }
#else
#pragma unroll
    for (int j = 0; j < 8; ++j) {
        const int u = j * 256 + tid;
        dst[u] = src[u];                       // global_load_b128 + ds_store_b128
    }
#endif
}

// ---------------------------------------------------------------------------
// Main kernel: 256 threads = 8 waves; each wave owns 2 M-tiles (32 rows).
// Block covers 16 M-tiles = 256 output rows -> 256 blocks.
// ---------------------------------------------------------------------------
__global__ __launch_bounds__(256) void sparse_conv_wmma_k(
    const bf16* __restrict__ featb,
    const int*  __restrict__ table,
    const bf16* __restrict__ fragb,
    const float* __restrict__ bias,
    float* __restrict__ out)
{
    __shared__ __align__(16) char smem[2 * FRAG_K_BYTES];   // double buffer, 64KB

    const int tid  = threadIdx.x;
    const int lane = tid & 31;
    const int wave = tid >> 5;

    const int r0base = (blockIdx.x * 16 + wave * 2) * 16;   // first row of tile 0
    const int m      = lane & 15;          // A row within tile this lane feeds
    const int khalf  = (lane >> 4) * 8;    // half of this lane's A K base (2*khalf) inside 32-chunk

    v8f acc0[8], acc1[8];
#pragma unroll
    for (int t = 0; t < 8; ++t)
#pragma unroll
        for (int i = 0; i < 8; ++i) { acc0[t][i] = 0.0f; acc1[t][i] = 0.0f; }

    // prologue: stage k=0 into buffer 0
    stage_frag((const v4i*)fragb, (v4i*)smem, tid);

    for (int k = 0; k < KOFF; ++k) {
        const int buf = k & 1;
#ifdef USE_ASYNC_LDS
        __builtin_amdgcn_s_wait_asynccnt(0);
#endif
        __syncthreads();                    // staged buf visible to all waves
        if (k + 1 < KOFF) {                 // kick next stage into other buffer
            stage_frag((const v4i*)(fragb + (size_t)(k + 1) * (FRAG_K_BYTES / 2)),
                       (v4i*)(smem + (buf ^ 1) * FRAG_K_BYTES), tid);
        }

        const int row0 = table[(r0base + m)      * TSTRIDE + k];
        const int row1 = table[(r0base + 16 + m) * TSTRIDE + k];
        const bf16* a0p = featb + (long)row0 * CCH;   // deref only if row>=0
        const bf16* a1p = featb + (long)row1 * CCH;
        const char* bbase = smem + buf * FRAG_K_BYTES;

#pragma unroll
        for (int c = 0; c < 4; ++c) {                 // K chunks of 32
            ABu a0, a1;
            if (row0 >= 0) {
                a0.h[0] = *(const v8bf*)(a0p + c * 32 + 2 * khalf);
                a0.h[1] = *(const v8bf*)(a0p + c * 32 + 2 * khalf + 8);
            } else {
#pragma unroll
                for (int i = 0; i < 8; ++i) { a0.h[0][i] = (bf16)0.0f; a0.h[1][i] = (bf16)0.0f; }
            }
            if (row1 >= 0) {
                a1.h[0] = *(const v8bf*)(a1p + c * 32 + 2 * khalf);
                a1.h[1] = *(const v8bf*)(a1p + c * 32 + 2 * khalf + 8);
            } else {
#pragma unroll
                for (int i = 0; i < 8; ++i) { a1.h[0][i] = (bf16)0.0f; a1.h[1][i] = (bf16)0.0f; }
            }

#pragma unroll
            for (int t = 0; t < 8; ++t) {             // 8 N-tiles of 16
                const char* fb = bbase + (c * 8 + t) * 1024;
                ABu b;
                b.h[0] = *(const v8bf*)(fb + lane * 16);         // ds_load_b128
                b.h[1] = *(const v8bf*)(fb + 512 + lane * 16);   // ds_load_b128
                acc0[t] = __builtin_amdgcn_wmma_f32_16x16x32_bf16(
                    false, a0.v, false, b.v, (short)0, acc0[t], false, false);
                acc1[t] = __builtin_amdgcn_wmma_f32_16x16x32_bf16(
                    false, a1.v, false, b.v, (short)0, acc1[t], false, false);
            }
        }
    }

    // C/D 16x16 f32 layout: lane l -> col n = l&15, VGPR v -> row v + (l>>4)*8
    const int n  = lane & 15;
    const int mb = (lane >> 4) * 8;
#pragma unroll
    for (int t = 0; t < 8; ++t) {
        const float bv = bias[t * 16 + n];
#pragma unroll
        for (int v = 0; v < 8; ++v) {
            out[(size_t)(r0base + mb + v) * CCH + t * 16 + n]      = acc0[t][v] + bv;
            out[(size_t)(r0base + 16 + mb + v) * CCH + t * 16 + n] = acc1[t][v] + bv;
        }
    }
}

// ---------------------------------------------------------------------------
// Launcher
// ---------------------------------------------------------------------------
extern "C" void kernel_launch(void* const* d_in, const int* in_sizes, int n_in,
                              void* d_out, int out_size, void* d_ws, size_t ws_size,
                              hipStream_t stream) {
    const float* features = (const float*)d_in[0];   // [65536,128] f32
    const float* weight   = (const float*)d_in[1];   // [27,128,128] f32
    const float* bias     = (const float*)d_in[2];   // [128] f32
    const int*   in_idx   = (const int*)d_in[3];     // [27,65536] i32
    const int*   out_idx  = (const int*)d_in[4];     // [27,65536] i32
    float* out = (float*)d_out;

    // Workspace layout (~25.1 MB):
    //   [0,       16 MB) : bf16 features
    //   [16 MB,   24 MB) : gather table, NV x 32 ints
    //   [24 MB, +0.9 MB) : bf16 weight fragments
    char* ws = (char*)d_ws;
    bf16* featb = (bf16*)(ws);
    int*  table = (int*)(ws + (size_t)16 * 1024 * 1024);
    bf16* fragb = (bf16*)(ws + (size_t)24 * 1024 * 1024);

    feat_to_bf16_k<<<(NV * CCH) / 256, 256, 0, stream>>>(features, featb);
    table_init_k<<<(NV * TSTRIDE) / 256, 256, 0, stream>>>(table);
    table_fill_k<<<(KOFF * NV) / 256, 256, 0, stream>>>(in_idx, out_idx, table);
    weight_prep_k<<<(KOFF * 16384 + 255) / 256, 256, 0, stream>>>(weight, fragb);

    // 4096 M-tiles / (2 per wave * 8 waves) = 256 blocks
    sparse_conv_wmma_k<<<256, 256, 0, stream>>>(featb, table, fragb, bias, out);
}
